// LargeGraphDualEncoder_944892805140
// MI455X (gfx1250) — compile-verified, matrix operand order checked by reading, not run on hardware
//
#include <hip/hip_runtime.h>
#include <hip/hip_bf16.h>
#include <stdint.h>

// ---------------------------------------------------------------------------
// Problem constants (from the reference)
// ---------------------------------------------------------------------------
#define GN 20000      // nodes
#define GE 320000     // edges
#define GP 100000     // pairs
#define GIN 768
#define GHID 256
#define GH 2
#define GD 256
#define GHD 512       // H*D

typedef __attribute__((ext_vector_type(16))) __bf16 v16bf;
typedef __attribute__((ext_vector_type(8)))  __bf16 v8bf;
typedef __attribute__((ext_vector_type(8)))  float  v8f;

// LDS tile geometry: 64 rows x 32 K bf16, padded to 40 elems/row (80B stride)
#define TPAD 40
#define TILE_ELEMS (64 * TPAD)

// ---------------------------------------------------------------------------
// Helpers
// ---------------------------------------------------------------------------
__device__ __forceinline__ __bf16 f2bf(float f) {
  unsigned int u = __builtin_bit_cast(unsigned int, f);
  unsigned int r = (u + 0x7FFFu + ((u >> 16) & 1u)) >> 16;   // RNE
  return __builtin_bit_cast(__bf16, (unsigned short)r);
}
// order-preserving float<->uint encoding for atomicMax on floats
__device__ __forceinline__ unsigned int fenc(float f) {
  unsigned int u = __builtin_bit_cast(unsigned int, f);
  return (u & 0x80000000u) ? ~u : (u | 0x80000000u);
}
__device__ __forceinline__ float fdec(unsigned int u) {
  u = (u & 0x80000000u) ? (u & 0x7FFFFFFFu) : ~u;
  return __builtin_bit_cast(float, u);
}
// assemble a 16-elem bf16 fragment from two 16B halves (p, p+16 elems)
__device__ __forceinline__ v16bf frag16(const __bf16* p) {
  v8bf lo = *(const v8bf*)p;
  v8bf hi = *(const v8bf*)(p + 16);
  return __builtin_shufflevector(lo, hi,
      0, 1, 2, 3, 4, 5, 6, 7, 8, 9, 10, 11, 12, 13, 14, 15);
}
// async 16B copy global -> LDS (CDNA5 GLOBAL_LOAD_ASYNC_TO_LDS_B128, ASYNCcnt)
__device__ __forceinline__ void async_copy_b128(unsigned lds_off, const __bf16* gptr) {
  asm volatile("global_load_async_to_lds_b128 %0, %1, off"
               :: "v"(lds_off), "v"(gptr) : "memory");
}
__device__ __forceinline__ void wait_async_le(int n) {
  if (n == 0) asm volatile("s_wait_asynccnt 0x0" ::: "memory");
  else        asm volatile("s_wait_asynccnt 0x2" ::: "memory");
}

// ---------------------------------------------------------------------------
// Packing kernels: f32 -> bf16 (row-major copy, and transposed weights)
// ---------------------------------------------------------------------------
__global__ void pack_bf16_kernel(const float* __restrict__ in,
                                 __bf16* __restrict__ out, long n) {
  long i = (long)blockIdx.x * blockDim.x + threadIdx.x;
  if (i < n) out[i] = f2bf(in[i]);
}

// W [K x Nout] f32 row-major  ->  WT [Nout x K] bf16 row-major
__global__ void pack_weight_T_kernel(const float* __restrict__ W,
                                     __bf16* __restrict__ WT,
                                     int K, int Nout) {
  long t = (long)blockIdx.x * blockDim.x + threadIdx.x;
  if (t >= (long)K * Nout) return;
  int n  = (int)(t / K);
  int kk = (int)(t % K);             // consecutive t -> consecutive k (coalesced write)
  WT[t] = f2bf(W[(size_t)kk * Nout + n]);
}

// ---------------------------------------------------------------------------
// bf16 WMMA GEMM with async-LDS double buffering.
//   A  : bf16 [M x K] row-major
//   WT : bf16 [Nout x K] row-major (W transposed, K contiguous)
//   C  : f32 [M x Nout] = act(A*W + bias)
// Block = 256 threads (8 waves) -> 64x64 output tile.
// Wave (mSub = w&3, nSub = w>>2) computes a 16x32 strip: 2 WMMAs per K-step
// sharing one A fragment.  Per 32-K step, A(64x32) + B(64x32) bf16 tiles are
// staged into LDS via global_load_async_to_lds_b128 (2 per thread),
// double-buffered with s_wait_asynccnt + barriers.
// Requires Nout % 64 == 0, K % 32 == 0 (rows of M are clamped/guarded).
// ---------------------------------------------------------------------------
__global__ void __launch_bounds__(256)
gemm_bf16_wmma(const __bf16* __restrict__ A,
               const __bf16* __restrict__ WT,
               const float* __restrict__ bias,
               float* __restrict__ C,
               int M, int Nout, int K, int do_relu) {
  __shared__ __bf16 tileA[2][TILE_ELEMS];
  __shared__ __bf16 tileB[2][TILE_ELEMS];

  const int tid  = threadIdx.x;
  const int wave = tid >> 5;
  const int lane = tid & 31;
  const int ro   = lane & 15;      // row-in-tile (A) / col-in-tile (B,C)
  const int half = lane >> 4;      // K-half selector (ISA 16-bit A 16x32 layout)
  const int mSub = wave & 3;       // 4 M sub-tiles of 16 rows
  const int nSub = wave >> 2;      // 2 N halves of 32 cols

  const int mBase = blockIdx.y * 64;
  const int nBase = blockIdx.x * 64;

  // copy assignment: thread t moves row (t>>2), 16B chunk (t&3) of each tile
  const int crow   = tid >> 2;         // 0..63
  const int cchunk = (tid & 3) * 8;    // element offset 0,8,16,24
  int gRowA = mBase + crow;
  if (gRowA > M - 1) gRowA = M - 1;    // clamp (stores are guarded)
  const __bf16* gA = A  + (size_t)gRowA * K + cchunk;
  const __bf16* gB = WT + (size_t)(nBase + crow) * K + cchunk;
  const unsigned ldsA0 = (unsigned)(uintptr_t)&tileA[0][crow * TPAD + cchunk];
  const unsigned ldsA1 = (unsigned)(uintptr_t)&tileA[1][crow * TPAD + cchunk];
  const unsigned ldsB0 = (unsigned)(uintptr_t)&tileB[0][crow * TPAD + cchunk];
  const unsigned ldsB1 = (unsigned)(uintptr_t)&tileB[1][crow * TPAD + cchunk];

  // fragment read pointers within a buffer
  const int aOff = (mSub * 16 + ro) * TPAD + half * 8;
  const int bOff = (nSub * 32 + ro) * TPAD + half * 8;

  const int nk = K >> 5;
  v8f acc0 = {}, acc1 = {};

  // prologue: stage k-tile 0 into buffer 0
  async_copy_b128(ldsA0, gA);
  async_copy_b128(ldsB0, gB);

  for (int kt = 0; kt < nk; ++kt) {
    const int cur = kt & 1;
    const bool more = (kt + 1) < nk;
    if (more) {   // stage next k-tile into the other buffer
      const int nxtk = (kt + 1) << 5;
      async_copy_b128(cur ? ldsA0 : ldsA1, gA + nxtk);
      async_copy_b128(cur ? ldsB0 : ldsB1, gB + nxtk);
    }
    wait_async_le(more ? 2 : 0);   // current buffer's copies complete
    __syncthreads();               // ... for every wave in the block

    const __bf16* Ab  = &tileA[cur][aOff];
    const __bf16* Bb0 = &tileB[cur][bOff];
    v16bf a  = frag16(Ab);
    v16bf b0 = frag16(Bb0);
    v16bf b1 = frag16(Bb0 + 16 * TPAD);
    acc0 = __builtin_amdgcn_wmma_f32_16x16x32_bf16(
        false, a, false, b0, (short)0, acc0, false, false);
    acc1 = __builtin_amdgcn_wmma_f32_16x16x32_bf16(
        false, a, false, b1, (short)0, acc1, false, false);
    __syncthreads();               // all reads done before buffer is re-filled
  }

  // C layout: VGPR r holds M = r + 8*half, N = lane&15
  const int cCol0 = nBase + nSub * 32 + ro;
  const float bv0 = bias ? bias[cCol0] : 0.0f;
  const float bv1 = bias ? bias[cCol0 + 16] : 0.0f;
#pragma unroll
  for (int r = 0; r < 8; ++r) {
    const int cRow = mBase + mSub * 16 + r + half * 8;
    if (cRow < M) {
      float v0 = acc0[r] + bv0;
      float v1 = acc1[r] + bv1;
      if (do_relu) { v0 = fmaxf(v0, 0.0f); v1 = fmaxf(v1, 0.0f); }
      C[(size_t)cRow * Nout + cCol0]      = v0;
      C[(size_t)cRow * Nout + cCol0 + 16] = v1;
    }
  }
}

// ---------------------------------------------------------------------------
// fill kernel (used to zero m_enc / s / agg)
// ---------------------------------------------------------------------------
__global__ void fill_u32(unsigned int* __restrict__ p, unsigned int val, long n) {
  long i = (long)blockIdx.x * blockDim.x + threadIdx.x;
  if (i < n) p[i] = val;
}

// ---------------------------------------------------------------------------
// edge logits + running segment max.  One wave per (edge, head).
// ---------------------------------------------------------------------------
__global__ void edge_logits_kernel(const float* __restrict__ q,
                                   const float* __restrict__ k,
                                   const long long* __restrict__ ei,
                                   float* __restrict__ logits,
                                   unsigned int* __restrict__ m_enc) {
  int task = blockIdx.x * (blockDim.x >> 5) + (threadIdx.x >> 5);
  if (task >= GE * GH) return;
  int e = task >> 1, h = task & 1;
  int lane = threadIdx.x & 31;
  long long src = ei[e];
  long long dst = ei[GE + e];
  const float* qrow = q + (size_t)dst * GHD + h * GD;
  const float* krow = k + (size_t)src * GHD + h * GD;
  float sum = 0.0f;
#pragma unroll
  for (int d = lane; d < GD; d += 32) sum += qrow[d] * krow[d];
#pragma unroll
  for (int off = 16; off > 0; off >>= 1) sum += __shfl_down(sum, off, 32);
  if (lane == 0) {
    float lg = sum * 0.0625f;   // 1/sqrt(256)
    logits[task] = lg;
    atomicMax(&m_enc[(size_t)dst * GH + h], fenc(lg));
  }
}

// p = exp(logit - m[dst]); s[dst] += p   (one thread per (e,h))
__global__ void edge_softmax_p_kernel(const unsigned int* __restrict__ m_enc,
                                      const long long* __restrict__ ei,
                                      float* __restrict__ logits,
                                      float* __restrict__ s) {
  int t = blockIdx.x * blockDim.x + threadIdx.x;
  if (t >= GE * GH) return;
  int e = t >> 1, h = t & 1;
  long long dst = ei[GE + e];
  float m = fdec(m_enc[(size_t)dst * GH + h]);
  float p = __expf(logits[t] - m);
  logits[t] = p;
  __hip_atomic_fetch_add(&s[(size_t)dst * GH + h], p,
                         __ATOMIC_RELAXED, __HIP_MEMORY_SCOPE_AGENT);
}

// agg[dst,h,:] += (p/s[dst,h]) * v[src,h,:]   (one wave per (e,h))
__global__ void edge_aggregate_kernel(const float* __restrict__ pvals,
                                      const float* __restrict__ s,
                                      const float* __restrict__ v,
                                      const long long* __restrict__ ei,
                                      float* __restrict__ agg) {
  int task = blockIdx.x * (blockDim.x >> 5) + (threadIdx.x >> 5);
  if (task >= GE * GH) return;
  int e = task >> 1, h = task & 1;
  int lane = threadIdx.x & 31;
  long long src = ei[e];
  long long dst = ei[GE + e];
  float coef = pvals[task] / s[(size_t)dst * GH + h];
  const float* vrow = v + (size_t)src * GHD + h * GD;
  float* arow = agg + (size_t)dst * GHD + h * GD;
#pragma unroll
  for (int d = lane; d < GD; d += 32)
    __hip_atomic_fetch_add(&arow[d], coef * vrow[d],
                           __ATOMIC_RELAXED, __HIP_MEMORY_SCOPE_AGENT);
}

// hout = relu(agg + skip)
__global__ void combine_relu_kernel(const float* __restrict__ agg,
                                    const float* __restrict__ skip,
                                    float* __restrict__ hout, long n) {
  long i = (long)blockIdx.x * blockDim.x + threadIdx.x;
  if (i < n) hout[i] = fmaxf(agg[i] + skip[i], 0.0f);
}

// pair features -> bf16 [P x 1024]:  [|vis-vjs|, vis*vjs, |via-vja|, via*vja]
__global__ void pair_feat_kernel(const float* __restrict__ zsyn,
                                 const float* __restrict__ zant,
                                 const long long* __restrict__ pe,
                                 __bf16* __restrict__ pf) {
  long t = (long)blockIdx.x * blockDim.x + threadIdx.x;
  if (t >= (long)GP * GHID) return;
  int p = (int)(t >> 8);
  int c = (int)(t & 255);
  long long i = pe[p];
  long long j = pe[GP + p];
  float vis = zsyn[(size_t)i * GHID + c], vjs = zsyn[(size_t)j * GHID + c];
  float via = zant[(size_t)i * GHID + c], vja = zant[(size_t)j * GHID + c];
  __bf16* row = pf + (size_t)p * 1024;
  row[c]       = f2bf(fabsf(vis - vjs));
  row[256 + c] = f2bf(vis * vjs);
  row[512 + c] = f2bf(fabsf(via - vja));
  row[768 + c] = f2bf(via * vja);
}

// out[p] = dot(ph[p,:], m2_W) + m2_b     (one wave per pair)
__global__ void pair_final_kernel(const float* __restrict__ ph,
                                  const float* __restrict__ m2W,
                                  const float* __restrict__ m2b,
                                  float* __restrict__ out) {
  int task = blockIdx.x * (blockDim.x >> 5) + (threadIdx.x >> 5);
  if (task >= GP) return;
  int lane = threadIdx.x & 31;
  const float* row = ph + (size_t)task * GHID;
  float sum = 0.0f;
#pragma unroll
  for (int d = lane; d < GHID; d += 32) sum += row[d] * m2W[d];
#pragma unroll
  for (int off = 16; off > 0; off >>= 1) sum += __shfl_down(sum, off, 32);
  if (lane == 0) out[task] = sum + m2b[0];
}

// ---------------------------------------------------------------------------
// Host-side orchestration
// ---------------------------------------------------------------------------
static void run_pack(const float* in, __bf16* out, long n, hipStream_t s) {
  pack_bf16_kernel<<<(unsigned)((n + 255) / 256), 256, 0, s>>>(in, out, n);
}
static void run_packT(const float* W, __bf16* WT, int K, int Nout, hipStream_t s) {
  long n = (long)K * Nout;
  pack_weight_T_kernel<<<(unsigned)((n + 255) / 256), 256, 0, s>>>(W, WT, K, Nout);
}
static void run_gemm(const __bf16* A, const __bf16* WT, const float* b,
                     float* C, int M, int Nout, int K, int relu, hipStream_t s) {
  dim3 grid(Nout / 64, (M + 63) / 64);
  gemm_bf16_wmma<<<grid, 256, 0, s>>>(A, WT, b, C, M, Nout, K, relu);
}

static void run_transformer_layer(const __bf16* hin_bf, int Kin,
                                  const long long* ei,
                                  const __bf16* WqT, const float* bq,
                                  const __bf16* WkT, const float* bk,
                                  const __bf16* WvT, const float* bv,
                                  const __bf16* WsT, const float* bs,
                                  float* q, float* k, float* v, float* skip,
                                  float* logits, unsigned int* m_enc, float* s,
                                  float* agg, float* hout, hipStream_t st) {
  run_gemm(hin_bf, WqT, bq, q,    GN, GHD, Kin, 0, st);
  run_gemm(hin_bf, WkT, bk, k,    GN, GHD, Kin, 0, st);
  run_gemm(hin_bf, WvT, bv, v,    GN, GHD, Kin, 0, st);
  run_gemm(hin_bf, WsT, bs, skip, GN, GHD, Kin, 0, st);

  long nmh = (long)GN * GH;
  fill_u32<<<(unsigned)((nmh + 255) / 256), 256, 0, st>>>(m_enc, 0u, nmh);
  fill_u32<<<(unsigned)((nmh + 255) / 256), 256, 0, st>>>((unsigned int*)s, 0u, nmh);
  long nagg = (long)GN * GHD;
  fill_u32<<<(unsigned)((nagg + 255) / 256), 256, 0, st>>>((unsigned int*)agg, 0u, nagg);

  int tasks = GE * GH;                         // one wave per task
  int wavesPerBlk = 256 / 32;
  int blk = (tasks + wavesPerBlk - 1) / wavesPerBlk;
  edge_logits_kernel<<<blk, 256, 0, st>>>(q, k, ei, logits, m_enc);
  edge_softmax_p_kernel<<<(tasks + 255) / 256, 256, 0, st>>>(m_enc, ei, logits, s);
  edge_aggregate_kernel<<<blk, 256, 0, st>>>(logits, s, v, ei, agg);

  combine_relu_kernel<<<(unsigned)((nagg + 255) / 256), 256, 0, st>>>(agg, skip, hout, nagg);
}

extern "C" void kernel_launch(void* const* d_in, const int* in_sizes, int n_in,
                              void* d_out, int out_size, void* d_ws, size_t ws_size,
                              hipStream_t stream) {
  (void)in_sizes; (void)n_in; (void)out_size; (void)ws_size;

  const float*     x    = (const float*)d_in[0];
  const long long* ei   = (const long long*)d_in[1];   // int64 [2,E]
  const long long* pe   = (const long long*)d_in[2];   // int64 [2,P]
  const float* Wq0 = (const float*)d_in[3];  const float* bq0 = (const float*)d_in[4];
  const float* Wk0 = (const float*)d_in[5];  const float* bk0 = (const float*)d_in[6];
  const float* Wv0 = (const float*)d_in[7];  const float* bv0 = (const float*)d_in[8];
  const float* Ws0 = (const float*)d_in[9];  const float* bs0 = (const float*)d_in[10];
  const float* Wq1 = (const float*)d_in[11]; const float* bq1 = (const float*)d_in[12];
  const float* Wk1 = (const float*)d_in[13]; const float* bk1 = (const float*)d_in[14];
  const float* Wv1 = (const float*)d_in[15]; const float* bv1 = (const float*)d_in[16];
  const float* Ws1 = (const float*)d_in[17]; const float* bs1 = (const float*)d_in[18];
  const float* postW = (const float*)d_in[19]; const float* postB = (const float*)d_in[20];
  const float* synW  = (const float*)d_in[21]; const float* synB  = (const float*)d_in[22];
  const float* antW  = (const float*)d_in[23]; const float* antB  = (const float*)d_in[24];
  const float* m1W   = (const float*)d_in[25]; const float* m1B   = (const float*)d_in[26];
  const float* m2W   = (const float*)d_in[27]; const float* m2B   = (const float*)d_in[28];
  float* out = (float*)d_out;

  // ---- workspace partition ----
  char* base = (char*)d_ws;
  size_t off = 0;
  auto wsalloc = [&](size_t bytes) -> void* {
    void* p = base + off;
    off = (off + bytes + 255) & ~(size_t)255;
    return p;
  };
  // f32 activations
  float* q    = (float*)wsalloc((size_t)GN * GHD * 4);
  float* k    = (float*)wsalloc((size_t)GN * GHD * 4);
  float* v    = (float*)wsalloc((size_t)GN * GHD * 4);
  float* skip = (float*)wsalloc((size_t)GN * GHD * 4);
  float* agg  = (float*)wsalloc((size_t)GN * GHD * 4);
  float* h0   = (float*)wsalloc((size_t)GN * GHD * 4);
  float* h1   = (float*)wsalloc((size_t)GN * GHD * 4);
  float* logits = (float*)wsalloc((size_t)GE * GH * 4);
  unsigned int* m_enc = (unsigned int*)wsalloc((size_t)GN * GH * 4);
  float* ssum = (float*)wsalloc((size_t)GN * GH * 4);
  float* hpost = (float*)wsalloc((size_t)GN * GHID * 4);
  float* zsyn  = (float*)wsalloc((size_t)GN * GHID * 4);
  float* zant  = (float*)wsalloc((size_t)GN * GHID * 4);
  float* ph    = (float*)wsalloc((size_t)GP * GHID * 4);
  // bf16 packed activations
  __bf16* xbf  = (__bf16*)wsalloc((size_t)GN * GIN * 2);
  __bf16* hbf  = (__bf16*)wsalloc((size_t)GN * GHD * 2);
  __bf16* hpbf = (__bf16*)wsalloc((size_t)GN * GHID * 2);
  __bf16* pf   = (__bf16*)wsalloc((size_t)GP * 1024 * 2);
  // bf16 transposed weights
  __bf16* Wq0T = (__bf16*)wsalloc((size_t)GIN * GHD * 2);
  __bf16* Wk0T = (__bf16*)wsalloc((size_t)GIN * GHD * 2);
  __bf16* Wv0T = (__bf16*)wsalloc((size_t)GIN * GHD * 2);
  __bf16* Ws0T = (__bf16*)wsalloc((size_t)GIN * GHD * 2);
  __bf16* Wq1T = (__bf16*)wsalloc((size_t)GHD * GHD * 2);
  __bf16* Wk1T = (__bf16*)wsalloc((size_t)GHD * GHD * 2);
  __bf16* Wv1T = (__bf16*)wsalloc((size_t)GHD * GHD * 2);
  __bf16* Ws1T = (__bf16*)wsalloc((size_t)GHD * GHD * 2);
  __bf16* postWT = (__bf16*)wsalloc((size_t)GHD * GHID * 2);
  __bf16* synWT  = (__bf16*)wsalloc((size_t)GHID * GHID * 2);
  __bf16* antWT  = (__bf16*)wsalloc((size_t)GHID * GHID * 2);
  __bf16* m1WT   = (__bf16*)wsalloc((size_t)(4 * GHID) * GHID * 2);

  // ---- pack all weights to transposed bf16 (once) ----
  run_packT(Wq0, Wq0T, GIN, GHD, stream);
  run_packT(Wk0, Wk0T, GIN, GHD, stream);
  run_packT(Wv0, Wv0T, GIN, GHD, stream);
  run_packT(Ws0, Ws0T, GIN, GHD, stream);
  run_packT(Wq1, Wq1T, GHD, GHD, stream);
  run_packT(Wk1, Wk1T, GHD, GHD, stream);
  run_packT(Wv1, Wv1T, GHD, GHD, stream);
  run_packT(Ws1, Ws1T, GHD, GHD, stream);
  run_packT(postW, postWT, GHD, GHID, stream);
  run_packT(synW, synWT, GHID, GHID, stream);
  run_packT(antW, antWT, GHID, GHID, stream);
  run_packT(m1W, m1WT, 4 * GHID, GHID, stream);

  // ---- layer 0 (K=768) ----
  run_pack(x, xbf, (long)GN * GIN, stream);
  run_transformer_layer(xbf, GIN, ei, Wq0T, bq0, Wk0T, bk0, Wv0T, bv0, Ws0T, bs0,
                        q, k, v, skip, logits, m_enc, ssum, agg, h0, stream);
  // ---- layer 1 (K=512) ----
  run_pack(h0, hbf, (long)GN * GHD, stream);
  run_transformer_layer(hbf, GHD, ei, Wq1T, bq1, Wk1T, bk1, Wv1T, bv1, Ws1T, bs1,
                        q, k, v, skip, logits, m_enc, ssum, agg, h1, stream);

  // ---- post / syn / ant MLPs ----
  run_pack(h1, hbf, (long)GN * GHD, stream);
  run_gemm(hbf, postWT, postB, hpost, GN, GHID, GHD, 1, stream);
  run_pack(hpost, hpbf, (long)GN * GHID, stream);
  run_gemm(hpbf, synWT, synB, zsyn, GN, GHID, GHID, 1, stream);
  run_gemm(hpbf, antWT, antB, zant, GN, GHID, GHID, 1, stream);

  // ---- pair features + pair MLP ----
  long nfeat = (long)GP * GHID;
  pair_feat_kernel<<<(unsigned)((nfeat + 255) / 256), 256, 0, stream>>>(zsyn, zant, pe, pf);
  run_gemm(pf, m1WT, m1B, ph, GP, GHID, 4 * GHID, 1, stream);

  int wavesPerBlk = 256 / 32;
  pair_final_kernel<<<(GP + wavesPerBlk - 1) / wavesPerBlk, 256, 0, stream>>>(ph, m2W, m2B, out);
}
